// Multi_head_focus_attention_4818953307055
// MI455X (gfx1250) — matrix operand order repaired and checked by reading, not compiled, not run on hardware
//
#include <hip/hip_runtime.h>
#include <hip/hip_bf16.h>

#define B_   16
#define C_   256
#define H_   128
#define W_   128
#define HW_  (H_*W_)
#define NH_  8
#define PAD_ 2

typedef __attribute__((ext_vector_type(16))) __bf16         v16bf;
typedef __attribute__((ext_vector_type(8)))  float          v8f;
typedef __attribute__((ext_vector_type(8)))  unsigned short ushort8;

__device__ __forceinline__ unsigned short f2bf(float f) {
    unsigned int u = __float_as_uint(f);
    return (unsigned short)((u + 0x7FFFu + ((u >> 16) & 1u)) >> 16);  // RNE
}
__device__ __forceinline__ float bf2f(unsigned short h) {
    return __uint_as_float(((unsigned int)h) << 16);
}
// Pack two floats into bf16x2 (one v_cvt_pk_bf16_f32 when available).
__device__ __forceinline__ unsigned int pk_bf16(float a, float b) {
#if defined(__has_builtin) && __has_builtin(__builtin_amdgcn_cvt_pk_bf16_f32)
    typedef __attribute__((ext_vector_type(2))) __bf16 v2bf;
    v2bf r = __builtin_amdgcn_cvt_pk_bf16_f32(a, b);
    union { v2bf v; unsigned int u; } cv; cv.v = r;
    return cv.u;
#else
    return (unsigned int)f2bf(a) | ((unsigned int)f2bf(b) << 16);
#endif
}
__device__ __forceinline__ v8f vzero8() {
    v8f z;
#pragma unroll
    for (int i = 0; i < 8; ++i) z[i] = 0.f;
    return z;
}
// Build a 16-element bf16 fragment from two 16B-aligned LDS chunks of 8 halves.
__device__ __forceinline__ v16bf frag_from(const unsigned short* p0, const unsigned short* p1) {
    ushort8 lo = *(const ushort8*)p0;
    ushort8 hi = *(const ushort8*)p1;
    v16bf f;
    union { unsigned short s; __bf16 b; } cv;
#pragma unroll
    for (int i = 0; i < 8; ++i) {
        cv.s = lo[i]; f[i]     = cv.b;
        cv.s = hi[i]; f[i + 8] = cv.b;
    }
    return f;
}

// ---------------------------------------------------------------------------
// Kernel 1: Y[256, B*HW] = W[256,256] * X[256, B*HW]   (per-batch channel GEMM)
// grid = (B*HW/128, 2), block = 256 (8 waves). Tile: 128 out-ch x 128 pixels.
// Double-buffered LDS staging + software-pipelined B-fragment loads.
// ---------------------------------------------------------------------------
__global__ __launch_bounds__(256)
void proj_gemm(const float* __restrict__ x, const float* __restrict__ w,
               unsigned short* __restrict__ y) {
    __shared__ unsigned short ldsW[2][128 * 40];  // [o][k] stride 40 halves
    __shared__ unsigned short ldsX[2][128 * 40];  // [pix][k] stride 40 halves

    const int tid    = threadIdx.x;
    const int lane   = tid & 31;
    const int wave   = tid >> 5;
    const int laneHi = lane >> 4;
    const int lane15 = lane & 15;

    const int pixGlobal = blockIdx.x * 128;
    const int b         = pixGlobal / HW_;
    const int pin       = pixGlobal % HW_;
    const int oBase     = blockIdx.y * 128;

    const float*    xb = x + (size_t)b * C_ * HW_ + pin;
    unsigned short* yb = y + (size_t)b * C_ * HW_ + pin;

    v8f acc[8];
#pragma unroll
    for (int i = 0; i < 8; ++i) acc[i] = vzero8();

    // ---- staging helpers (pairs along k; packed b32 LDS stores) ----
    auto stageW = [&](int buf, int kc) {
#pragma unroll
        for (int it = 0; it < 8; ++it) {
            int j  = tid + it * 256;        // pair index, 2048 pairs = 128x16
            int o  = j >> 4;
            int kp = (j & 15) * 2;
            float2 wv = *(const float2*)(w + (size_t)(oBase + o) * C_ + kc + kp);
            *(unsigned int*)&ldsW[buf][o * 40 + kp] = pk_bf16(wv.x, wv.y);
        }
    };
    auto stageX = [&](int buf, int kc) {
#pragma unroll
        for (int it = 0; it < 8; ++it) {
            int j  = tid + it * 256;        // pair index, 2048 pairs = 128px x 16
            int k2 = (j >> 7) * 2;
            int p  = j & 127;
            float a0 = xb[(size_t)(kc + k2) * HW_ + p];
            float a1 = xb[(size_t)(kc + k2 + 1) * HW_ + p];
            *(unsigned int*)&ldsX[buf][p * 40 + k2] = pk_bf16(a0, a1);
        }
    };

    stageW(0, 0);
    stageX(0, 0);

    for (int i = 0; i < 8; ++i) {
        __syncthreads();
        const int cur = i & 1;
        if (i < 7) {
            stageW(cur ^ 1, (i + 1) * 32);
            stageX(cur ^ 1, (i + 1) * 32);
        }
        const unsigned short* abase = &ldsW[cur][(wave * 16 + lane15) * 40 + laneHi * 8];
        v16bf afrag = frag_from(abase, abase + 16);
        const unsigned short* bbase = &ldsX[cur][lane15 * 40 + laneHi * 8];
        v16bf bnext = frag_from(bbase, bbase + 16);  // preload tile 0
#pragma unroll
        for (int nt = 0; nt < 8; ++nt) {
            v16bf bcur = bnext;
            if (nt < 7) {                            // preload tile nt+1 before WMMA nt
                const unsigned short* br = bbase + (nt + 1) * (16 * 40);
                bnext = frag_from(br, br + 16);
            }
            acc[nt] = __builtin_amdgcn_wmma_f32_16x16x32_bf16(
                false, afrag, false, bcur, (short)0, acc[nt], false, false);
        }
    }

    const int o0 = oBase + wave * 16 + laneHi * 8;
#pragma unroll
    for (int nt = 0; nt < 8; ++nt) {
        int p = nt * 16 + lane15;
#pragma unroll
        for (int r2 = 0; r2 < 4; ++r2) {
            unsigned int pk = pk_bf16(acc[nt][2 * r2], acc[nt][2 * r2 + 1]);
            yb[(size_t)(o0 + 2 * r2)     * HW_ + p] = (unsigned short)pk;
            yb[(size_t)(o0 + 2 * r2 + 1) * HW_ + p] = (unsigned short)(pk >> 16);
        }
    }
}

// ---------------------------------------------------------------------------
// Kernel 2: 5x5 local-window attention per (b, head, 16x16 pixel tile).
// grid = B*NH*64, block = 256 (one thread per pixel).
// ---------------------------------------------------------------------------
__global__ __launch_bounds__(256)
void local_attn(const unsigned short* __restrict__ qh,
                const unsigned short* __restrict__ kh,
                const unsigned short* __restrict__ vh,
                unsigned short* __restrict__ oh) {
    __shared__ unsigned short sk[400 * 40];  // padded 20x20 tile, [y*20+x][c] stride 40
    __shared__ unsigned short sv[400 * 40];

    const int tid  = threadIdx.x;
    const int bid  = blockIdx.x;
    const int tile = bid & 63;
    const int head = (bid >> 6) & 7;
    const int b    = bid >> 9;
    const int tx   = (tile & 7) * 16;
    const int ty   = (tile >> 3) * 16;
    const size_t base = (size_t)b * C_ * HW_ + (size_t)head * 32 * HW_;

    // stage zero-padded k/v tiles (20x20 spatial, 32 channels)
    for (int i = tid; i < 32 * 400; i += 256) {
        int c  = i / 400;
        int r  = i - c * 400;
        int ky = r / 20, kx = r - ky * 20;
        int gy = ty + ky - PAD_, gx = tx + kx - PAD_;
        unsigned short kvl = 0, vvl = 0;
        if (gy >= 0 && gy < H_ && gx >= 0 && gx < W_) {
            size_t g = base + (size_t)c * HW_ + (size_t)gy * W_ + gx;
            kvl = kh[g];
            vvl = vh[g];
        }
        sk[r * 40 + c] = kvl;
        sv[r * 40 + c] = vvl;
    }
    __syncthreads();

    const int py = tid >> 4, px = tid & 15;

    // this thread's q vector (32 channels)
    float qc[32];
    {
        size_t g = base + (size_t)(ty + py) * W_ + tx + px;
#pragma unroll
        for (int c = 0; c < 32; ++c) qc[c] = bf2f(qh[g + (size_t)c * HW_]);
    }

    // correlation over the 25 window offsets
    float corr[25];
#pragma unroll
    for (int wdx = 0; wdx < 25; ++wdx) {
        int dy = wdx / 5, dx = wdx - dy * 5;
        const unsigned short* kp = &sk[((py + dy) * 20 + (px + dx)) * 40];
        float a = 0.f;
#pragma unroll
        for (int cc = 0; cc < 4; ++cc) {
            uint4 u = *(const uint4*)(kp + cc * 8);
            a += qc[cc * 8 + 0] * __uint_as_float(u.x << 16);
            a += qc[cc * 8 + 1] * __uint_as_float(u.x & 0xffff0000u);
            a += qc[cc * 8 + 2] * __uint_as_float(u.y << 16);
            a += qc[cc * 8 + 3] * __uint_as_float(u.y & 0xffff0000u);
            a += qc[cc * 8 + 4] * __uint_as_float(u.z << 16);
            a += qc[cc * 8 + 5] * __uint_as_float(u.z & 0xffff0000u);
            a += qc[cc * 8 + 6] * __uint_as_float(u.w << 16);
            a += qc[cc * 8 + 7] * __uint_as_float(u.w & 0xffff0000u);
        }
        corr[wdx] = a;
    }

    // relu -> /sqrt(32) -> softmax over 25 (OOB slots contribute exp(0))
    const float scale = 0.17677669529663687f;  // 1/sqrt(32)
    float m = 0.f;
#pragma unroll
    for (int wdx = 0; wdx < 25; ++wdx) {
        float cval = corr[wdx] > 0.f ? corr[wdx] * scale : 0.f;
        corr[wdx]  = cval;
        m = fmaxf(m, cval);
    }
    float s = 0.f;
#pragma unroll
    for (int wdx = 0; wdx < 25; ++wdx) { float e = __expf(corr[wdx] - m); corr[wdx] = e; s += e; }
    float inv = 1.0f / s;
#pragma unroll
    for (int wdx = 0; wdx < 25; ++wdx) corr[wdx] *= inv;

    // aggregate v over the window
    float ao[32];
#pragma unroll
    for (int c = 0; c < 32; ++c) ao[c] = 0.f;
#pragma unroll
    for (int wdx = 0; wdx < 25; ++wdx) {
        int dy = wdx / 5, dx = wdx - dy * 5;
        float a = corr[wdx];
        const unsigned short* vp = &sv[((py + dy) * 20 + (px + dx)) * 40];
#pragma unroll
        for (int cc = 0; cc < 4; ++cc) {
            uint4 u = *(const uint4*)(vp + cc * 8);
            ao[cc * 8 + 0] += a * __uint_as_float(u.x << 16);
            ao[cc * 8 + 1] += a * __uint_as_float(u.x & 0xffff0000u);
            ao[cc * 8 + 2] += a * __uint_as_float(u.y << 16);
            ao[cc * 8 + 3] += a * __uint_as_float(u.y & 0xffff0000u);
            ao[cc * 8 + 4] += a * __uint_as_float(u.z << 16);
            ao[cc * 8 + 5] += a * __uint_as_float(u.z & 0xffff0000u);
            ao[cc * 8 + 6] += a * __uint_as_float(u.w << 16);
            ao[cc * 8 + 7] += a * __uint_as_float(u.w & 0xffff0000u);
        }
    }

    size_t g = base + (size_t)(ty + py) * W_ + tx + px;
#pragma unroll
    for (int c2 = 0; c2 < 16; ++c2) {
        unsigned int pk = pk_bf16(ao[2 * c2], ao[2 * c2 + 1]);
        oh[g + (size_t)(2 * c2)     * HW_] = (unsigned short)pk;
        oh[g + (size_t)(2 * c2 + 1) * HW_] = (unsigned short)(pk >> 16);
    }
}

// ---------------------------------------------------------------------------
// Kernel 3: o = wfc * out + residual, then per-pixel channel LayerNorm.
// grid = B*HW/64, block = 256 (8 waves). Tile: all 256 out-ch x 64 pixels.
// ---------------------------------------------------------------------------
__global__ __launch_bounds__(256)
void fc_ln(const unsigned short* __restrict__ oh, const float* __restrict__ wfc,
           const float* __restrict__ resid, const float* __restrict__ lnw,
           const float* __restrict__ lnb, float* __restrict__ out) {
    __shared__ unsigned short ldsW[2][256 * 40];
    __shared__ unsigned short ldsX[2][64 * 40];
    __shared__ float          sO[256 * 65];   // stride 65 -> conflict-free column reads
    __shared__ float          sRed[512];      // [0..255]=sum, [256..511]=sumsq
    __shared__ float          sStat[128];     // [0..63]=mu, [64..127]=rstd

    const int tid    = threadIdx.x;
    const int lane   = tid & 31;
    const int wave   = tid >> 5;
    const int laneHi = lane >> 4;
    const int lane15 = lane & 15;

    const int pixGlobal = blockIdx.x * 64;
    const int b         = pixGlobal / HW_;
    const int pin       = pixGlobal % HW_;
    const unsigned short* xb   = oh + (size_t)b * C_ * HW_ + pin;
    const float*          resb = resid + (size_t)b * C_ * HW_ + pin;

    // prefetch the residual tile (256 rows x 64 px = 2 cachelines/row)
    {
        const char* pr = (const char*)(resb + (size_t)tid * HW_);
        __builtin_prefetch(pr, 0, 1);
        __builtin_prefetch(pr + 128, 0, 1);
    }

    v8f acc[2][4];
#pragma unroll
    for (int i = 0; i < 2; ++i)
#pragma unroll
        for (int j = 0; j < 4; ++j) acc[i][j] = vzero8();

    auto stageW = [&](int buf, int kc) {
#pragma unroll
        for (int it = 0; it < 16; ++it) {
            int j  = tid + it * 256;        // pair index, 4096 pairs = 256x16
            int o  = j >> 4;
            int kp = (j & 15) * 2;
            float2 wv = *(const float2*)(wfc + (size_t)o * C_ + kc + kp);
            *(unsigned int*)&ldsW[buf][o * 40 + kp] = pk_bf16(wv.x, wv.y);
        }
    };
    auto stageX = [&](int buf, int kc) {
#pragma unroll
        for (int it = 0; it < 8; ++it) {
            int i = tid + it * 256;         // 2048 elements = 32k x 64p
            int k = i >> 6, p = i & 63;
            ldsX[buf][p * 40 + k] = xb[(size_t)(kc + k) * HW_ + p];  // already bf16
        }
    };

    stageW(0, 0);
    stageX(0, 0);

    for (int i = 0; i < 8; ++i) {
        __syncthreads();
        const int cur = i & 1;
        if (i < 7) {
            stageW(cur ^ 1, (i + 1) * 32);
            stageX(cur ^ 1, (i + 1) * 32);
        }
        // hoist all fragments: 2 A-tiles + 4 shared B-tiles, then 8 WMMAs
        v16bf af[2], bf[4];
#pragma unroll
        for (int mi = 0; mi < 2; ++mi) {
            const unsigned short* arow = &ldsW[cur][((wave * 2 + mi) * 16 + lane15) * 40 + laneHi * 8];
            af[mi] = frag_from(arow, arow + 16);
        }
#pragma unroll
        for (int nt = 0; nt < 4; ++nt) {
            const unsigned short* brow = &ldsX[cur][(nt * 16 + lane15) * 40 + laneHi * 8];
            bf[nt] = frag_from(brow, brow + 16);
        }
#pragma unroll
        for (int mi = 0; mi < 2; ++mi)
#pragma unroll
            for (int nt = 0; nt < 4; ++nt)
                acc[mi][nt] = __builtin_amdgcn_wmma_f32_16x16x32_bf16(
                    false, af[mi], false, bf[nt], (short)0, acc[mi][nt], false, false);
    }
    __syncthreads();

    // scatter GEMM result into f32 LDS tile
#pragma unroll
    for (int mi = 0; mi < 2; ++mi)
#pragma unroll
        for (int nt = 0; nt < 4; ++nt)
#pragma unroll
            for (int r = 0; r < 8; ++r) {
                int oc = (wave * 2 + mi) * 16 + r + laneHi * 8;
                int p  = nt * 16 + lane15;
                sO[oc * 65 + p] = acc[mi][nt][r];
            }
    __syncthreads();

    // residual add + partial channel sums: 4 threads per pixel, 64 channels each
    const int p  = tid & 63;
    const int qq = tid >> 6;
    {
        float sum = 0.f, sq = 0.f;
        for (int c = qq * 64; c < qq * 64 + 64; ++c) {
            float val = sO[c * 65 + p] + resb[(size_t)c * HW_ + p];
            sO[c * 65 + p] = val;
            sum += val;
            sq  += val * val;
        }
        sRed[tid]       = sum;
        sRed[256 + tid] = sq;
    }
    __syncthreads();
    if (tid < 64) {
        float su  = sRed[p] + sRed[p + 64] + sRed[p + 128] + sRed[p + 192];
        float sqq = sRed[256 + p] + sRed[320 + p] + sRed[384 + p] + sRed[448 + p];
        float mu  = su * (1.f / 256.f);
        float var = sqq * (1.f / 256.f) - mu * mu;
        sStat[p]      = mu;
        sStat[64 + p] = rsqrtf(var + 1e-6f);
    }
    __syncthreads();
    {
        float mu   = sStat[p];
        float rstd = sStat[64 + p];
        float* ob = out + (size_t)b * C_ * HW_ + pin + p;
        for (int c = qq * 64; c < qq * 64 + 64; ++c) {
            ob[(size_t)c * HW_] = lnw[c] * ((sO[c * 65 + p] - mu) * rstd) + lnb[c];
        }
    }
}

// ---------------------------------------------------------------------------
extern "C" void kernel_launch(void* const* d_in, const int* in_sizes, int n_in,
                              void* d_out, int out_size, void* d_ws, size_t ws_size,
                              hipStream_t stream) {
    const float* q   = (const float*)d_in[0];
    const float* k   = (const float*)d_in[1];
    const float* v   = (const float*)d_in[2];
    const float* wq  = (const float*)d_in[3];
    const float* wk  = (const float*)d_in[4];
    const float* wv  = (const float*)d_in[5];
    const float* wfc = (const float*)d_in[6];
    const float* lnw = (const float*)d_in[7];
    const float* lnb = (const float*)d_in[8];
    float* out = (float*)d_out;

    const size_t elems = (size_t)B_ * C_ * HW_;  // 67,108,864 per tensor
    unsigned short* qh = (unsigned short*)d_ws;  // bf16 intermediates in workspace
    unsigned short* kh = qh + elems;
    unsigned short* vh = kh + elems;
    unsigned short* oh = vh + elems;             // total 4 * 128 MiB = 512 MiB

    dim3 blk(256);
    dim3 gP(B_ * HW_ / 128, 2);
    proj_gemm<<<gP, blk, 0, stream>>>(q, wq, qh);
    proj_gemm<<<gP, blk, 0, stream>>>(k, wk, kh);
    proj_gemm<<<gP, blk, 0, stream>>>(v, wv, vh);
    local_attn<<<dim3(B_ * NH_ * 64), blk, 0, stream>>>(qh, kh, vh, oh);
    fc_ln<<<dim3(B_ * HW_ / 64), blk, 0, stream>>>(oh, wfc, q, lnw, lnb, out);
}
